// BahdanauAttention_44547400794489
// MI455X (gfx1250) — compile-verified, matrix operand order checked by reading, not run on hardware
//
#include <hip/hip_runtime.h>
#include <stdint.h>

#define B_ 64
#define T_ 2048
#define D_ 512
#define U_ 512
#define ROWS_ 32   // t-rows per block (2 WMMA M-tiles)

typedef __attribute__((ext_vector_type(16))) __bf16 v16bf;
typedef __attribute__((ext_vector_type(8)))  float  v8f;
typedef __attribute__((ext_vector_type(4)))  unsigned int v4u;
typedef __attribute__((ext_vector_type(8)))  int v8i_t;
typedef __attribute__((ext_vector_type(4)))  int v4i_t;

__device__ __forceinline__ unsigned short f32_to_bf16_rne(float f) {
    unsigned int u = __float_as_uint(f);
    unsigned int r = u + 0x7FFFu + ((u >> 16) & 1u);
    return (unsigned short)(r >> 16);
}

// CDNA5 hardware tanh (TRANS32 op). Two v_nops cover the TRANS RAW hazard,
// since the compiler cannot see through the asm block.
__device__ __forceinline__ float tanh_hw(float x) {
    float y;
    asm("v_tanh_f32 %0, %1\n\tv_nop\n\tv_nop" : "=v"(y) : "v"(x));
    return y;
}

// ---------------- Pass 0a: qp[b,u] = query[b,:] @ W1[:,u] + b1[u] ----------------
__global__ void k_prep_qp(const float* __restrict__ query, const float* __restrict__ W1,
                          const float* __restrict__ b1, float* __restrict__ qp) {
    int idx = blockIdx.x * blockDim.x + threadIdx.x;
    if (idx >= B_ * U_) return;
    int b = idx / U_, u = idx % U_;
    float acc = b1[u];
    const float* q = query + (size_t)b * D_;
    #pragma unroll 8
    for (int k = 0; k < D_; ++k) acc += q[k] * W1[(size_t)k * U_ + u];
    qp[idx] = acc;
}

// ---------------- Pass 0b: W2t[u,d] = bf16(W2[d,u]) ----------------
__global__ void k_prep_w2t(const float* __restrict__ W2, unsigned short* __restrict__ W2t) {
    int idx = blockIdx.x * blockDim.x + threadIdx.x;
    if (idx >= U_ * D_) return;
    int u = idx / D_, d = idx % D_;
    W2t[idx] = f32_to_bf16_rne(W2[(size_t)d * U_ + u]);
}

// ---------------- Pass 1: score[b,t] = tanh(qp[b]+values[b,t]@W2+b2) @ V + bV ----------------
// grid (T/32, B), 512 threads = 16 waves. Each wave: 2 N-tiles (32 U-cols) x 2 M-tiles (32 t-rows).
__global__ __launch_bounds__(512, 2)
void k_score(const float* __restrict__ values, const unsigned short* __restrict__ W2t,
             const float* __restrict__ qp, const float* __restrict__ b2,
             const float* __restrict__ Vw, const float* __restrict__ bV,
             float* __restrict__ score) {
    __shared__ __align__(16) float          Af[ROWS_ * D_];   // TDM destination (f32 tile, 64KB)
    __shared__ __align__(16) unsigned short Abf[ROWS_ * D_];  // bf16-converted tile (32KB)
    __shared__ float sscore[ROWS_];

    const int b    = blockIdx.y;
    const int t0   = blockIdx.x * ROWS_;
    const int tid  = threadIdx.x;
    const int wave = tid >> 5;
    const int lane = tid & 31;
    const int half = lane >> 4;   // 0: lanes 0-15, 1: lanes 16-31
    const int lrow = lane & 15;

    if (tid < ROWS_) sscore[tid] = 0.0f;

    // --- Tensor Data Mover: DMA the 32x512 f32 values tile into LDS (wave 0 only) ---
    if (wave == 0) {
        unsigned long long ga =
            (unsigned long long)(uintptr_t)(values + ((size_t)b * T_ + t0) * (size_t)D_);
        unsigned int lds_off = (unsigned int)(uintptr_t)(&Af[0]);

        v4u g0;
        g0[0] = 1u;                                     // count = 1 valid descriptor
        g0[1] = lds_off;                                // lds_addr
        g0[2] = (unsigned int)ga;                       // global_addr[31:0]
        g0[3] = ((unsigned int)(ga >> 32) & 0x01FFFFFFu) | (2u << 30);  // addr[56:32] | type=2

        v8i_t g1;
        g1[0] = (int)(2u << 16);          // data_size = 2 (4 bytes)
        g1[1] = (int)(512u << 16);        // tensor_dim0 = 512 (D, contiguous)
        g1[2] = (int)(2048u << 16);       // tensor_dim1 = 2048 (T)
        g1[3] = (int)(512u << 16);        // tile_dim0 = 512
        g1[4] = ROWS_;                    // tile_dim1 = 32 rows
        g1[5] = 512;                      // tensor_dim0_stride = 512
        g1[6] = 0;
        g1[7] = 0;
        v4i_t gz  = {0, 0, 0, 0};
        v8i_t gz8 = {0, 0, 0, 0, 0, 0, 0, 0};
        __builtin_amdgcn_tensor_load_to_lds(g0, g1, gz, gz, gz8, 0);
        __builtin_amdgcn_s_wait_tensorcnt(0);
    }
    __syncthreads();

    // --- cooperative f32 -> bf16 conversion of the A tile ---
    for (int i = tid; i < ROWS_ * D_; i += 512) Abf[i] = f32_to_bf16_rne(Af[i]);
    __syncthreads();

    const int nb = wave * 32;             // this wave's first U column (2 tiles of 16)

    v8f acc00 = {0.f,0.f,0.f,0.f,0.f,0.f,0.f,0.f};   // M-tile 0, N-tile 0
    v8f acc01 = acc00;                               // M-tile 0, N-tile 1
    v8f acc10 = acc00;                               // M-tile 1, N-tile 0
    v8f acc11 = acc00;                               // M-tile 1, N-tile 1

    union Frag { v16bf v; uint4 q[2]; };

    const uint4* Ap0 = reinterpret_cast<const uint4*>(&Abf[lrow * D_]);          // rows 0-15
    const uint4* Ap1 = reinterpret_cast<const uint4*>(&Abf[(16 + lrow) * D_]);   // rows 16-31
    const uint4* B0  = reinterpret_cast<const uint4*>(W2t + (size_t)(nb + lrow) * D_);
    const uint4* B1  = reinterpret_cast<const uint4*>(W2t + (size_t)(nb + 16 + lrow) * D_);

    #pragma unroll 2
    for (int k0 = 0; k0 < D_; k0 += 32) {
        // ISA 16-bit fragment layout: lanes 0-15 hold K 0-7 & 16-23, lanes 16-31 hold K 8-15 & 24-31
        const int kLo = k0 + half * 8;        // uint4 index = K/8 (8 bf16 = 16 bytes)
        const int kHi = k0 + 16 + half * 8;

        Frag a0, a1, fb0, fb1;
        a0.q[0]  = Ap0[kLo >> 3]; a0.q[1]  = Ap0[kHi >> 3];
        a1.q[0]  = Ap1[kLo >> 3]; a1.q[1]  = Ap1[kHi >> 3];
        fb0.q[0] = B0[kLo >> 3];  fb0.q[1] = B0[kHi >> 3];
        fb1.q[0] = B1[kLo >> 3];  fb1.q[1] = B1[kHi >> 3];

        acc00 = __builtin_amdgcn_wmma_f32_16x16x32_bf16(false, a0.v, false, fb0.v,
                                                        (short)0, acc00, false, false);
        acc01 = __builtin_amdgcn_wmma_f32_16x16x32_bf16(false, a0.v, false, fb1.v,
                                                        (short)0, acc01, false, false);
        acc10 = __builtin_amdgcn_wmma_f32_16x16x32_bf16(false, a1.v, false, fb0.v,
                                                        (short)0, acc10, false, false);
        acc11 = __builtin_amdgcn_wmma_f32_16x16x32_bf16(false, a1.v, false, fb1.v,
                                                        (short)0, acc11, false, false);
    }

    // --- epilogue: tanh(acc + qp + b2) * V, reduce over N ---
    float p0[8], p1[8];
    #pragma unroll
    for (int r = 0; r < 8; ++r) { p0[r] = 0.f; p1[r] = 0.f; }

    #pragma unroll
    for (int tile = 0; tile < 2; ++tile) {
        const int n = nb + tile * 16 + lrow;       // this lane's column within the tile
        const float qb = qp[b * U_ + n] + b2[n];
        const float vv = Vw[n];
        v8f am0 = tile ? acc01 : acc00;
        v8f am1 = tile ? acc11 : acc10;
        #pragma unroll
        for (int r = 0; r < 8; ++r) {              // C layout: VGPR r -> M = r + 8*half
            p0[r] += tanh_hw(am0[r] + qb) * vv;
            p1[r] += tanh_hw(am1[r] + qb) * vv;
        }
    }

    // reduce across the 16 lanes of each half (masks < 16 keep halves separate)
    #pragma unroll
    for (int m = 1; m < 16; m <<= 1) {
        #pragma unroll
        for (int r = 0; r < 8; ++r) {
            p0[r] += __shfl_xor(p0[r], m, 32);
            p1[r] += __shfl_xor(p1[r], m, 32);
        }
    }
    if (lrow == 0) {
        #pragma unroll
        for (int r = 0; r < 8; ++r) {
            atomicAdd(&sscore[r + half * 8],      p0[r]);
            atomicAdd(&sscore[16 + r + half * 8], p1[r]);
        }
    }
    __syncthreads();

    if (tid < ROWS_) score[(size_t)b * T_ + t0 + tid] = sscore[tid] + bV[0];
}

// ---------------- Pass 2: softmax over T, then context = sum_t attn * values ----------------
__global__ __launch_bounds__(512)
void k_softmax_ctx(const float* __restrict__ score, const float* __restrict__ values,
                   float* __restrict__ ctx, float* __restrict__ attn) {
    __shared__ float probs[T_];
    __shared__ float red[16];

    const int b = blockIdx.x, tid = threadIdx.x;
    const int wave = tid >> 5, lane = tid & 31;
    const float* sc = score + (size_t)b * T_;

    // max over T
    float m = -3.402823466e38f;
    for (int t = tid; t < T_; t += 512) m = fmaxf(m, sc[t]);
    #pragma unroll
    for (int s = 16; s >= 1; s >>= 1) m = fmaxf(m, __shfl_xor(m, s, 32));
    if (lane == 0) red[wave] = m;
    __syncthreads();
    float M = red[0];
    #pragma unroll
    for (int i = 1; i < 16; ++i) M = fmaxf(M, red[i]);
    __syncthreads();

    // exp + sum
    float s = 0.f;
    for (int t = tid; t < T_; t += 512) {
        float e = __expf(sc[t] - M);
        probs[t] = e;
        s += e;
    }
    #pragma unroll
    for (int sh = 16; sh >= 1; sh >>= 1) s += __shfl_xor(s, sh, 32);
    if (lane == 0) red[wave] = s;
    __syncthreads();
    float S = 0.f;
    #pragma unroll
    for (int i = 0; i < 16; ++i) S += red[i];
    const float inv = 1.0f / S;

    for (int t = tid; t < T_; t += 512) {
        float a = probs[t] * inv;
        probs[t] = a;
        attn[(size_t)b * T_ + t] = a;
    }
    __syncthreads();

    // context: thread d accumulates over T (coalesced 2KB rows, probs from LDS)
    const int d = tid;  // 512 threads == D_
    const float* vb = values + (size_t)b * T_ * D_;
    float acc = 0.f;
    #pragma unroll 8
    for (int t = 0; t < T_; ++t) acc += probs[t] * vb[(size_t)t * D_ + d];
    ctx[(size_t)b * D_ + d] = acc;
}

// ---------------- host-side launcher ----------------
extern "C" void kernel_launch(void* const* d_in, const int* in_sizes, int n_in,
                              void* d_out, int out_size, void* d_ws, size_t ws_size,
                              hipStream_t stream) {
    const float* query  = (const float*)d_in[0];
    const float* values = (const float*)d_in[1];
    const float* W1     = (const float*)d_in[2];
    const float* b1     = (const float*)d_in[3];
    const float* W2     = (const float*)d_in[4];
    const float* b2     = (const float*)d_in[5];
    const float* V      = (const float*)d_in[6];
    const float* bV     = (const float*)d_in[7];

    float* out  = (float*)d_out;
    float* ctx  = out;              // [B, D]
    float* attn = out + B_ * D_;    // [B, T]

    char* ws = (char*)d_ws;
    float*          qp    = (float*)ws;                                // 128 KB
    unsigned short* W2t   = (unsigned short*)(ws + 131072);            // 512 KB
    float*          score = (float*)(ws + 131072 + 524288);            // 512 KB

    k_prep_qp <<<(B_ * U_ + 255) / 256, 256, 0, stream>>>(query, W1, b1, qp);
    k_prep_w2t<<<(U_ * D_ + 255) / 256, 256, 0, stream>>>(W2, W2t);

    dim3 g2(T_ / ROWS_, B_);
    k_score<<<g2, 512, 0, stream>>>(values, W2t, qp, b2, V, bV, score);

    k_softmax_ctx<<<B_, 512, 0, stream>>>(score, values, ctx, attn);
}
